// TemporalFeatureExtractor_31834297598431
// MI455X (gfx1250) — compile-verified, matrix operand order checked by reading, not run on hardware
//
#include <hip/hip_runtime.h>
#include <hip/hip_bf16.h>
#include <math.h>

// Problem constants
#define BATCH 32
#define TLEN  512
#define INCH  512
#define HIDN  512
#define G4    2048            // gates per direction
#define GALL  4096            // both directions
#define MROWS (BATCH * TLEN)  // 16384 GEMM rows
#define NBLK  64              // persistent recurrence blocks (2 dirs x 32 slices)

typedef __bf16 bf16_t;
typedef bf16_t v16bf __attribute__((ext_vector_type(16)));
typedef float  v8f   __attribute__((ext_vector_type(8)));

union FragU { uint4 u[2]; v16bf v; };
union AccU  { v8f v; float f[8]; };

__device__ __forceinline__ unsigned short f2bf(float f) {
  unsigned int u = __float_as_uint(f);
  u += 0x7FFFu + ((u >> 16) & 1u);   // round-to-nearest-even
  return (unsigned short)(u >> 16);
}
__device__ __forceinline__ float sigm(float x) { return 1.0f / (1.0f + __expf(-x)); }

// ---------------------------------------------------------------------------
// K0: zero h0 (bf16 double-buffer slot 0, both dirs), c state, grid barrier
// ---------------------------------------------------------------------------
__global__ void init_state(unsigned short* __restrict__ hbf0, float* __restrict__ cbuf,
                           unsigned int* __restrict__ bar) {
  int tid = blockIdx.x * blockDim.x + threadIdx.x;   // 32768 threads
  hbf0[tid] = 0;
  cbuf[tid] = 0.0f;
  if (tid < 2) bar[tid] = 0u;
}

// ---------------------------------------------------------------------------
// K1: x (B,C,H,T) f32  ->  X (B*T, 512) bf16, i = c*16 + h
// ---------------------------------------------------------------------------
__global__ void pack_x(const float* __restrict__ x, unsigned short* __restrict__ xbf) {
  int tid = blockIdx.x * blockDim.x + threadIdx.x;   // 8388608 threads
  int i  = tid & 511;
  int bt = tid >> 9;
  int t  = bt & 511;
  int b  = bt >> 9;
  float v = x[((size_t)(b * 512 + i)) * 512 + t];
  xbf[tid] = f2bf(v);                                // xbf[(b*512+t)*512 + i]
}

// ---------------------------------------------------------------------------
// K2: weights -> bf16 (fwd rows 0..2047, bwd rows 2048..4095), bias sums
// ---------------------------------------------------------------------------
__global__ void pack_w(const float* __restrict__ wihf, const float* __restrict__ whhf,
                       const float* __restrict__ wihb, const float* __restrict__ whhb,
                       const float* __restrict__ bihf, const float* __restrict__ bhhf,
                       const float* __restrict__ bihb, const float* __restrict__ bhhb,
                       unsigned short* __restrict__ wih, unsigned short* __restrict__ whh,
                       float* __restrict__ bias) {
  int tid = blockIdx.x * blockDim.x + threadIdx.x;   // 4096*512 threads
  int g  = tid >> 9;
  int i  = tid & 511;
  int gl = g & 2047;
  bool back = g >= G4;
  const float* wi = back ? wihb : wihf;
  const float* wh = back ? whhb : whhf;
  wih[tid] = f2bf(wi[(size_t)gl * 512 + i]);
  whh[tid] = f2bf(wh[(size_t)gl * 512 + i]);
  if (tid < GALL) {
    bias[tid] = (tid >= G4) ? (bihb[tid - G4] + bhhb[tid - G4])
                            : (bihf[tid] + bhhf[tid]);
  }
}

// ---------------------------------------------------------------------------
// K3: xp[dir][m][g] = X(m,:) . W_ih(g,:) + bias[g]    (m = b*512 + t)
// grid: (1024 M-tiles, 8 N-groups of 512 cols), 256 threads = 8 waves.
// Each wave owns a 16x64 output strip: ONE A fragment per K-chunk is reused
// against 4 B tiles -> 4 independent WMMA chains, 64 WMMAs per wave total.
// Cuts A-side L1/L2 re-fetch traffic 4x vs. one-tile-per-wave.
// ---------------------------------------------------------------------------
__global__ void xp_gemm(const unsigned short* __restrict__ xbf,
                        const unsigned short* __restrict__ wih,
                        const float* __restrict__ bias,
                        float* __restrict__ xp) {
  int lane = threadIdx.x & 31;
  int wave = threadIdx.x >> 5;
  int m0 = blockIdx.x * 16;
  int G0 = blockIdx.y * 512 + wave * 64;   // 4 N-tiles: G0, +16, +32, +48
  int ln   = lane & 15;
  int kb   = (lane >> 4) * 8;   // K sub-offset per ISA A/B fragment layout
  int rsel = (lane >> 4) * 8;   // M sub-offset per ISA C/D layout

  const unsigned short* arow = xbf + (size_t)(m0 + ln) * INCH;
  const unsigned short* brow = wih + (size_t)(G0 + ln) * INCH;

  AccU acc[4];
#pragma unroll
  for (int j = 0; j < 4; ++j)
    acc[j].v = (v8f){0.f, 0.f, 0.f, 0.f, 0.f, 0.f, 0.f, 0.f};

#pragma unroll 2
  for (int k0 = 0; k0 < INCH; k0 += 32) {
    FragU a;
    a.u[0] = *(const uint4*)(arow + k0 + kb);
    a.u[1] = *(const uint4*)(arow + k0 + kb + 16);
#pragma unroll
    for (int j = 0; j < 4; ++j) {
      FragU b;
      b.u[0] = *(const uint4*)(brow + (size_t)j * 16 * INCH + k0 + kb);
      b.u[1] = *(const uint4*)(brow + (size_t)j * 16 * INCH + k0 + kb + 16);
      acc[j].v = __builtin_amdgcn_wmma_f32_16x16x32_bf16(
          false, a.v, false, b.v, (short)0, acc[j].v, false, false);
    }
  }

  int dir = G0 >> 11;                      // 64-col strip never crosses dirs
#pragma unroll
  for (int j = 0; j < 4; ++j) {
    int Gj = G0 + j * 16;
    float bs = bias[Gj + ln];
    int g = (Gj + ln) & 2047;
#pragma unroll
    for (int r = 0; r < 8; ++r) {
      int m = m0 + r + rsel;
      xp[((size_t)dir * MROWS + m) * G4 + g] = acc[j].f[r] + bs;
    }
  }
}

// ---------------------------------------------------------------------------
// K4: persistent bidirectional LSTM recurrence, one launch for all 512 steps.
// 64 blocks = 2 dirs x 32 j-slices of 16 h-columns; 256 threads = 8 waves.
// Block preloads its 64 W_hh gate-rows (64 KB bf16) into LDS ONCE; per step:
//   wave (mi = w&1 batch tile, q = w>>1 gate type): acc init from xp tile,
//   16 WMMAs (A: h from global/L2, B: W_hh from LDS via ds_load_b128),
//   gates staged in LDS, pointwise c/h update, h -> out + bf16 double buffer,
//   then a software grid barrier (atomic arrive + generation flag).
// ---------------------------------------------------------------------------
__global__ void lstm_persist(const unsigned short* __restrict__ whh,
                             const float* __restrict__ xp,
                             unsigned short* __restrict__ hbf,
                             float* __restrict__ cbuf,
                             float* __restrict__ out,
                             unsigned int* __restrict__ bar) {
  __shared__ unsigned short wsl[64 * HIDN];   // 64 KB: this block's W_hh rows
  __shared__ float gbuf[4][BATCH][16];        // 8 KB: gate pre-activations

  const int blk = blockIdx.x;
  const int dir = blk >> 5;
  const int j0  = (blk & 31) * 16;

  const int lane = threadIdx.x & 31;
  const int wave = threadIdx.x >> 5;
  const int mi   = wave & 1;          // batch tile (rows mi*16..mi*16+15)
  const int q    = wave >> 1;         // gate type: 0=i 1=f 2=g 3=o
  const int ln   = lane & 15;
  const int kb   = (lane >> 4) * 8;   // K sub-offset per ISA fragment layout
  const int rsel = (lane >> 4) * 8;   // M sub-offset per ISA C/D layout

  // --- one-time W_hh slice preload: rows (q,jj) -> local row q*16+jj -------
  for (int idx = threadIdx.x; idx < 64 * (HIDN / 8); idx += 256) {
    int row = idx >> 6;               // 0..63 local gate row
    int k8  = (idx & 63) * 8;         // element offset (16B chunks)
    int qq  = row >> 4, jj = row & 15;
    const unsigned short* src =
        whh + (size_t)(dir * G4 + qq * HIDN + j0 + jj) * HIDN + k8;
    *(uint4*)&wsl[row * HIDN + k8] = *(const uint4*)src;
  }
  __syncthreads();

  const unsigned short* bsrc = &wsl[(q * 16 + ln) * HIDN];
  const size_t HBUF = (size_t)2 * BATCH * HIDN;          // elems per h buffer
  const int gcol = q * HIDN + j0;

  for (int s = 0; s < TLEN; ++s) {
    const int t = dir ? (TLEN - 1 - s) : s;
    const unsigned short* hin_d =
        hbf + (size_t)(s & 1) * HBUF + (size_t)dir * (BATCH * HIDN);
    unsigned short* hout_d =
        hbf + (size_t)((s + 1) & 1) * HBUF + (size_t)dir * (BATCH * HIDN);
    const unsigned short* arow = hin_d + (size_t)(mi * 16 + ln) * HIDN;

    AccU acc;
#pragma unroll
    for (int r = 0; r < 8; ++r) {
      int b = mi * 16 + r + rsel;
      acc.f[r] = xp[((size_t)dir * MROWS + b * TLEN + t) * G4 + (gcol + ln)];
    }

#pragma unroll 2
    for (int k0 = 0; k0 < HIDN; k0 += 32) {
      FragU a, b;
      a.u[0] = *(const uint4*)(arow + k0 + kb);
      a.u[1] = *(const uint4*)(arow + k0 + kb + 16);
      b.u[0] = *(const uint4*)(bsrc + k0 + kb);        // ds_load_b128
      b.u[1] = *(const uint4*)(bsrc + k0 + kb + 16);
      acc.v = __builtin_amdgcn_wmma_f32_16x16x32_bf16(
          false, a.v, false, b.v, (short)0, acc.v, false, false);
    }

#pragma unroll
    for (int r = 0; r < 8; ++r)
      gbuf[q][mi * 16 + r + rsel][ln] = acc.f[r];

    __syncthreads();

#pragma unroll
    for (int e = threadIdx.x; e < BATCH * 16; e += 256) {
      int b  = e >> 4;
      int jj = e & 15;
      int j  = j0 + jj;
      float iv = gbuf[0][b][jj];
      float fv = gbuf[1][b][jj];
      float gv = gbuf[2][b][jj];
      float ov = gbuf[3][b][jj];
      size_t cidx = ((size_t)dir * BATCH + b) * HIDN + j;
      float c  = cbuf[cidx];
      float cn = sigm(fv) * c + sigm(iv) * tanhf(gv);
      float hn = sigm(ov) * tanhf(cn);
      cbuf[cidx] = cn;
      hout_d[(size_t)b * HIDN + j] = f2bf(hn);
      out[((size_t)b * 1024 + (size_t)dir * 512 + j) * TLEN + t] = hn;
    }

    // ---- software grid barrier across the 64 persistent blocks ------------
    __syncthreads();                       // all gbuf reads + h stores issued
    if (threadIdx.x == 0) {
      __threadfence();                     // make h stores visible at device scope
      unsigned int want = (unsigned int)(s + 1) * (unsigned int)NBLK;
      unsigned int arrived = atomicAdd(&bar[0], 1u) + 1u;
      if (arrived == want) {
        atomicExch(&bar[1], want);         // release this generation
      } else {
        volatile unsigned int* gen = &bar[1];
        while (*gen < want) __builtin_amdgcn_s_sleep(1);
      }
      __threadfence();                     // acquire before next step's h loads
    }
    __syncthreads();
  }
}

// ---------------------------------------------------------------------------
// Host-side orchestration
// ---------------------------------------------------------------------------
extern "C" void kernel_launch(void* const* d_in, const int* in_sizes, int n_in,
                              void* d_out, int out_size, void* d_ws, size_t ws_size,
                              hipStream_t stream) {
  const float* x     = (const float*)d_in[0];
  const float* wihf  = (const float*)d_in[1];
  const float* whhf  = (const float*)d_in[2];
  const float* bihf  = (const float*)d_in[3];
  const float* bhhf  = (const float*)d_in[4];
  const float* wihb  = (const float*)d_in[5];
  const float* whhb  = (const float*)d_in[6];
  const float* bihb  = (const float*)d_in[7];
  const float* bhhb  = (const float*)d_in[8];
  float* out = (float*)d_out;

  // Workspace layout (bytes)
  char* ws = (char*)d_ws;
  const size_t XBF_OFF  = 0;                                   // 16 MiB
  const size_t WIH_OFF  = XBF_OFF + (size_t)MROWS * INCH * 2;  // 4 MiB
  const size_t WHH_OFF  = WIH_OFF + (size_t)GALL * INCH * 2;   // 4 MiB
  const size_t BIAS_OFF = WHH_OFF + (size_t)GALL * HIDN * 2;   // 16 KiB
  const size_t XP_OFF   = BIAS_OFF + (size_t)GALL * 4;         // 256 MiB
  const size_t HBF_OFF  = XP_OFF + (size_t)2 * MROWS * G4 * 4; // 128 KiB
  const size_t CB_OFF   = HBF_OFF + (size_t)2 * 2 * BATCH * HIDN * 2;
  const size_t BAR_OFF  = CB_OFF + (size_t)2 * BATCH * HIDN * 4;

  unsigned short* xbf  = (unsigned short*)(ws + XBF_OFF);
  unsigned short* wih  = (unsigned short*)(ws + WIH_OFF);
  unsigned short* whh  = (unsigned short*)(ws + WHH_OFF);
  float*          bias = (float*)(ws + BIAS_OFF);
  float*          xp   = (float*)(ws + XP_OFF);
  unsigned short* hbf  = (unsigned short*)(ws + HBF_OFF);
  float*          cbuf = (float*)(ws + CB_OFF);
  unsigned int*   bar  = (unsigned int*)(ws + BAR_OFF);

  // 1. zero h0 / c0 / barrier state (re-done every call: deterministic)
  init_state<<<128, 256, 0, stream>>>(hbf, cbuf, bar);

  // 2. pack activations and weights to bf16
  pack_x<<<(BATCH * TLEN * INCH) / 256, 256, 0, stream>>>(x, xbf);
  pack_w<<<((size_t)GALL * INCH) / 256, 256, 0, stream>>>(
      wihf, whhf, wihb, whhb, bihf, bhhf, bihb, bhhb, wih, whh, bias);

  // 3. big input-projection GEMM (both directions), bias fused, N-blocked
  xp_gemm<<<dim3(MROWS / 16, GALL / 512), 256, 0, stream>>>(xbf, wih, bias, xp);

  // 4. persistent recurrence: single launch, 512 steps, grid-barrier synced
  lstm_persist<<<NBLK, 256, 0, stream>>>(whh, xp, hbf, cbuf, out, bar);
}
